// Decoupled_Rope_MHA_20899310862923
// MI455X (gfx1250) — compile-verified, matrix operand order checked by reading, not run on hardware
//
#include <hip/hip_runtime.h>
#include <hip/hip_bf16.h>
#include <math.h>

// ---------------------------------------------------------------------------
// Decoupled-RoPE causal MHA forward for MI455X (gfx1250, wave32, WMMA + TDM).
// cvt weights -> QKV GEMM (TDM-staged LDS, bf16 WMMA) -> RoPE + V-transpose ->
// flash attention (TDM-staged K/V tiles, bf16 WMMA, online softmax) -> out GEMM
// ---------------------------------------------------------------------------

#define D_MODEL   2048
#define N_HEADS   16
#define HEAD_DIM  128
#define ROPE_DIM  64
#define BATCH     4
#define SEQ       2048
#define ROWS      (BATCH * SEQ)        // 8192
#define QKV_N     (3 * D_MODEL)        // 6144

typedef __attribute__((ext_vector_type(16))) __bf16 v16bf;
typedef __attribute__((ext_vector_type(8)))  __bf16 v8bf;
typedef __attribute__((ext_vector_type(8)))  float  v8f;
typedef __attribute__((ext_vector_type(4)))  float  v4f;
typedef __attribute__((ext_vector_type(4)))  unsigned int u32x4;
typedef __attribute__((ext_vector_type(8)))  int    i32x8;
typedef __attribute__((ext_vector_type(4)))  int    i32x4;

#if defined(__AMDGCN__) && __has_builtin(__builtin_amdgcn_tensor_load_to_lds)
#define HAVE_TDM 1
#else
#define HAVE_TDM 0
#endif

static __device__ inline v8f zero8() {
    v8f z;
#pragma unroll
    for (int i = 0; i < 8; ++i) z[i] = 0.0f;
    return z;
}

static __device__ inline v8f wmma_bf16(v16bf a, v16bf b, v8f c) {
    return __builtin_amdgcn_wmma_f32_16x16x32_bf16(
        false, a, false, b, (short)0, c, false, false);
}

// A/B fragment loader from row-major [row, k] bf16 storage (global or LDS).
// ISA layout: lane%16 = row; element i -> k = (i%8) + 8*(lane/16) + 16*(i/8).
static __device__ inline v16bf load_frag(const __bf16* base, int ld,
                                         int row0, int k0, int lane) {
    int row = row0 + (lane & 15);
    int col = k0 + ((lane >> 4) << 3);
    const __bf16* p = base + (size_t)row * ld + col;
    v8bf lo = *(const v8bf*)(p);
    v8bf hi = *(const v8bf*)(p + 16);
    v16bf r;
#pragma unroll
    for (int i = 0; i < 8; ++i) { r[i] = lo[i]; r[8 + i] = hi[i]; }
    return r;
}

// Same, but f32 source converted to bf16 while building the fragment.
static __device__ inline v16bf load_frag(const float* base, int ld,
                                         int row0, int k0, int lane) {
    int row = row0 + (lane & 15);
    int col = k0 + ((lane >> 4) << 3);
    const float* p = base + (size_t)row * ld + col;
    v4f a0 = *(const v4f*)(p);
    v4f a1 = *(const v4f*)(p + 4);
    v4f b0 = *(const v4f*)(p + 16);
    v4f b1 = *(const v4f*)(p + 20);
    v16bf r;
#pragma unroll
    for (int i = 0; i < 4; ++i) {
        r[i]      = (__bf16)a0[i];
        r[4 + i]  = (__bf16)a1[i];
        r[8 + i]  = (__bf16)b0[i];
        r[12 + i] = (__bf16)b1[i];
    }
    return r;
}

// ---------------------------------------------------------------------------
// Tensor Data Mover: 2-D tile (global -> LDS).  D# packing per ISA 08 §8.3/8.4.
// ---------------------------------------------------------------------------
__device__ inline void tdm_load_2d(unsigned lds_off, unsigned long long gaddr,
                                   unsigned dsz_code,
                                   unsigned td0, unsigned td1,
                                   unsigned tile0, unsigned tile1,
                                   unsigned long long stride0) {
#if HAVE_TDM
    u32x4 g0;
    g0[0] = 1u;                                   // count=1, user mode, no gather
    g0[1] = lds_off;                              // LDS byte address
    g0[2] = (unsigned)(gaddr & 0xFFFFFFFFull);    // global addr [31:0]
    g0[3] = (unsigned)((gaddr >> 32) & 0x01FFFFFFull) | (2u << 30);  // type=2
    i32x8 g1;
    g1[0] = (int)(dsz_code << 16);                // wg_mask=0, data_size
    g1[1] = (int)((td0 & 0xFFFFu) << 16);         // tensor_dim0 [15:0]
    g1[2] = (int)(((td0 >> 16) & 0xFFFFu) | ((td1 & 0xFFFFu) << 16));
    g1[3] = (int)(((td1 >> 16) & 0xFFFFu) | ((tile0 & 0xFFFFu) << 16));
    g1[4] = (int)(tile1 & 0xFFFFu);               // tile_dim1, tile_dim2=0
    g1[5] = (int)(unsigned)(stride0 & 0xFFFFFFFFull);
    g1[6] = (int)((stride0 >> 32) & 0xFFFFull);   // dim1_stride=0
    g1[7] = 0;
    i32x4 z4;
#pragma unroll
    for (int i = 0; i < 4; ++i) z4[i] = 0;
#if defined(__clang_major__) && (__clang_major__ >= 23)
    i32x8 z8;
#pragma unroll
    for (int i = 0; i < 8; ++i) z8[i] = 0;
    __builtin_amdgcn_tensor_load_to_lds(g0, g1, z4, z4, z8, 0);
#else
    __builtin_amdgcn_tensor_load_to_lds(g0, g1, z4, z4, 0);
#endif
#else
    (void)lds_off; (void)gaddr; (void)dsz_code; (void)td0; (void)td1;
    (void)tile0; (void)tile1; (void)stride0;
#endif
}

template <typename T>
__device__ inline unsigned lds_off_of(const T* p) {
    return (unsigned)(size_t)(const void*)p;   // LDS aperture: offset in [31:0]
}

// Fallback cooperative tile stage (all 256 threads) when TDM is unavailable.
template <typename T>
__device__ inline void stage_tile(T* dst, const T* src, int rows, int cols,
                                  int ld, int tid) {
    for (int i = tid; i < rows * cols; i += 256) {
        int r = i / cols, c = i - r * cols;
        dst[i] = src[(size_t)r * ld + c];
    }
}

// ---------------------------------------------------------------------------
// f32 -> bf16 conversion (weights)
// ---------------------------------------------------------------------------
__global__ void cvt_f32_to_bf16(const float* __restrict__ in,
                                __bf16* __restrict__ out, int n) {
    for (int i = blockIdx.x * blockDim.x + threadIdx.x; i < n;
         i += gridDim.x * blockDim.x)
        out[i] = (__bf16)in[i];
}

// ---------------------------------------------------------------------------
// GEMM: C[M,N] = A[M,K] * B[N,K]^T  (bf16 WMMA, f32 accum)
// 256 thr = 8 waves; block tile 64(M) x 256(N); per wave 32x64.
// A (64x32) + B (256x32) k-slabs double-buffered in LDS via TDM (wave 0).
// MODE 0: bf16 scatter into qkv (t,b,h,s,d).  MODE 1: f32 row-major.
// ---------------------------------------------------------------------------
template <typename AT, int MODE>
__global__ __launch_bounds__(256) void gemm_wmma(const AT* __restrict__ A,
                                                 const __bf16* __restrict__ B,
                                                 void* __restrict__ Cout,
                                                 int M, int N, int K) {
    __shared__ __align__(16) char   AtileRaw[2][64 * 32 * sizeof(AT)];
    __shared__ __align__(16) __bf16 Btile[2][256 * 32];

    const int tid  = threadIdx.x;
    const int lane = tid & 31;
    const int w    = tid >> 5;
    const int row0 = blockIdx.y * 64;
    const int col0 = blockIdx.x * 256;
    const int wmL  = (w >> 2) * 32;   // wave M offset within block tile
    const int wnL  = (w & 3) * 64;    // wave N offset within block tile
    const unsigned dszA = (sizeof(AT) == 4) ? 2u : 1u;  // 4B / 2B elements

    v8f acc[2][4];
#pragma unroll
    for (int i = 0; i < 2; ++i)
#pragma unroll
        for (int j = 0; j < 4; ++j) acc[i][j] = zero8();

    const int nsteps = K / 32;
#if HAVE_TDM
    if (w == 0) {
        tdm_load_2d(lds_off_of(&AtileRaw[0][0]),
                    (unsigned long long)(size_t)(A + (size_t)row0 * K),
                    dszA, 32, 64, 32, 64, (unsigned)K);
        tdm_load_2d(lds_off_of(&Btile[0][0]),
                    (unsigned long long)(size_t)(B + (size_t)col0 * K),
                    1u, 32, 256, 32, 256, (unsigned)K);
    }
#endif
    for (int i = 0; i < nsteps; ++i) {
        const int p = i & 1;
#if HAVE_TDM
        if (w == 0) {
            if (i + 1 < nsteps) {
                const int kk = (i + 1) * 32;
                const int q = (i + 1) & 1;
                tdm_load_2d(lds_off_of(&AtileRaw[q][0]),
                            (unsigned long long)(size_t)(A + (size_t)row0 * K + kk),
                            dszA, 32, 64, 32, 64, (unsigned)K);
                tdm_load_2d(lds_off_of(&Btile[q][0]),
                            (unsigned long long)(size_t)(B + (size_t)col0 * K + kk),
                            1u, 32, 256, 32, 256, (unsigned)K);
                __builtin_amdgcn_s_wait_tensorcnt(2);
            } else {
                __builtin_amdgcn_s_wait_tensorcnt(0);
            }
        }
        __syncthreads();
#else
        __syncthreads();
        stage_tile((AT*)AtileRaw[p], A + (size_t)row0 * K + i * 32, 64, 32, K, tid);
        stage_tile(Btile[p], B + (size_t)col0 * K + i * 32, 256, 32, K, tid);
        __syncthreads();
#endif
        // preload all fragments, then one WMMA burst (better clause/co-exec)
        const AT* At = (const AT*)AtileRaw[p];
        v16bf a0 = load_frag(At, 32, wmL,      0, lane);
        v16bf a1 = load_frag(At, 32, wmL + 16, 0, lane);
        v16bf bfr[4];
#pragma unroll
        for (int j = 0; j < 4; ++j)
            bfr[j] = load_frag(Btile[p], 32, wnL + 16 * j, 0, lane);
#pragma unroll
        for (int j = 0; j < 4; ++j) {
            acc[0][j] = wmma_bf16(a0, bfr[j], acc[0][j]);
            acc[1][j] = wmma_bf16(a1, bfr[j], acc[1][j]);
        }
        __syncthreads();
    }

    const int hl = lane >> 4, ln = lane & 15;
#pragma unroll
    for (int i = 0; i < 2; ++i)
#pragma unroll
        for (int j = 0; j < 4; ++j)
#pragma unroll
            for (int r = 0; r < 8; ++r) {
                int mrow = row0 + wmL + 16 * i + r + 8 * hl;
                int ncol = col0 + wnL + 16 * j + ln;
                float v = acc[i][j][r];
                if (MODE == 0) {
                    int b = mrow >> 11, s = mrow & (SEQ - 1);
                    int t = ncol >> 11, hn = ncol & (D_MODEL - 1);
                    int h = hn >> 7,  d = hn & (HEAD_DIM - 1);
                    size_t idx = ((((size_t)t * BATCH + b) * N_HEADS + h) * SEQ + s)
                                 * HEAD_DIM + d;
                    ((__bf16*)Cout)[idx] = (__bf16)v;
                } else {
                    ((float*)Cout)[(size_t)mrow * N + ncol] = v;
                }
            }
}

// ---------------------------------------------------------------------------
// RoPE on last 64 dims of q,k; V transposed to (b,h,d,s).
// Q is pre-scaled by 1/sqrt(HEAD_DIM) so the flash inner loop skips scaling.
// ---------------------------------------------------------------------------
__global__ __launch_bounds__(HEAD_DIM) void rope_split_kernel(
    const __bf16* __restrict__ qkv,   // (t,b,h,s,d)
    const float* __restrict__ cosc, const float* __restrict__ sinc,
    __bf16* __restrict__ qo, __bf16* __restrict__ ko, __bf16* __restrict__ vt) {
    const int bh = blockIdx.x;
    const int s  = blockIdx.y;
    const int d  = threadIdx.x;
    const size_t plane = (size_t)BATCH * N_HEADS * SEQ * HEAD_DIM;
    const size_t rowQ  = ((size_t)bh * SEQ + s) * HEAD_DIM;
    const float sc = 0.08838834764831845f;   // 1/sqrt(128)

    const __bf16* q = qkv + rowQ;
    const __bf16* k = qkv + plane + rowQ;
    const __bf16* v = qkv + 2 * plane + rowQ;

    float qv, kv;
    if (d < ROPE_DIM) {
        qv = (float)q[d];
        kv = (float)k[d];
    } else if (d < ROPE_DIM + 32) {
        int j = d - ROPE_DIM;
        float c = cosc[s * 32 + j], sn = sinc[s * 32 + j];
        qv = (float)q[ROPE_DIM + 2 * j] * c - (float)q[ROPE_DIM + 2 * j + 1] * sn;
        kv = (float)k[ROPE_DIM + 2 * j] * c - (float)k[ROPE_DIM + 2 * j + 1] * sn;
    } else {
        int j = d - ROPE_DIM - 32;
        float c = cosc[s * 32 + j], sn = sinc[s * 32 + j];
        qv = (float)q[ROPE_DIM + 2 * j + 1] * c + (float)q[ROPE_DIM + 2 * j] * sn;
        kv = (float)k[ROPE_DIM + 2 * j + 1] * c + (float)k[ROPE_DIM + 2 * j] * sn;
    }
    qo[rowQ + d] = (__bf16)(qv * sc);
    ko[rowQ + d] = (__bf16)kv;
    vt[((size_t)bh * HEAD_DIM + d) * SEQ + s] = v[d];
}

// ---------------------------------------------------------------------------
// Flash attention, causal. grid (B*H, SEQ/128), 256 thr (8 waves).
// Block owns 128 Q rows (16 per wave); K (32x128) and V^T (128x32) tiles are
// TDM-staged into double-buffered LDS and shared by all waves. Waves past
// their causal range skip compute (wave-uniform; EXEC stays full for WMMA).
// Interior tiles (kb+31 <= qbase) skip masking entirely (wave-uniform).
// ---------------------------------------------------------------------------
__global__ __launch_bounds__(256) void flash_attn_kernel(
    const __bf16* __restrict__ Q,   // (b,h,s,d), pre-scaled by 1/sqrt(d)
    const __bf16* __restrict__ Kk,  // (b,h,s,d)
    const __bf16* __restrict__ Vt,  // (b,h,d,s)
    __bf16* __restrict__ Aout) {    // (b,s,D_MODEL)
    __shared__ __align__(16) __bf16 Ktile[2][32 * HEAD_DIM];
    __shared__ __align__(16) __bf16 Vtile[2][HEAD_DIM * 32];
    __shared__ __align__(16) __bf16 pbuf[8][16 * 32];

    const int tid  = threadIdx.x;
    const int lane = tid & 31;
    const int w    = tid >> 5;
    const int hl   = lane >> 4, ln = lane & 15;
    const int bh   = blockIdx.x;
    const int qb0  = blockIdx.y * 128;
    const int qbase = qb0 + w * 16;

    const __bf16* Qb = Q  + (size_t)bh * SEQ * HEAD_DIM;
    const __bf16* Kb = Kk + (size_t)bh * SEQ * HEAD_DIM;
    const __bf16* Vb = Vt + (size_t)bh * HEAD_DIM * SEQ;

    v16bf qf[4];
#pragma unroll
    for (int c = 0; c < 4; ++c) qf[c] = load_frag(Qb, HEAD_DIM, qbase, 32 * c, lane);

    v8f o[8];
#pragma unroll
    for (int c = 0; c < 8; ++c) o[c] = zero8();
    float mrow[8], lrow[8];
#pragma unroll
    for (int r = 0; r < 8; ++r) { mrow[r] = -3e38f; lrow[r] = 0.0f; }

    const int nsteps = (qb0 + 128) / 32;

#if HAVE_TDM
    if (w == 0) {
        tdm_load_2d(lds_off_of(&Ktile[0][0]),
                    (unsigned long long)(size_t)Kb,
                    1u, HEAD_DIM, 32, HEAD_DIM, 32, HEAD_DIM);
        tdm_load_2d(lds_off_of(&Vtile[0][0]),
                    (unsigned long long)(size_t)Vb,
                    1u, 32, HEAD_DIM, 32, HEAD_DIM, SEQ);
    }
#endif
    for (int i = 0; i < nsteps; ++i) {
        const int p = i & 1;
        const int kb = i * 32;
#if HAVE_TDM
        if (w == 0) {
            if (i + 1 < nsteps) {
                const int kn = (i + 1) * 32;
                const int q = (i + 1) & 1;
                tdm_load_2d(lds_off_of(&Ktile[q][0]),
                            (unsigned long long)(size_t)(Kb + (size_t)kn * HEAD_DIM),
                            1u, HEAD_DIM, 32, HEAD_DIM, 32, HEAD_DIM);
                tdm_load_2d(lds_off_of(&Vtile[q][0]),
                            (unsigned long long)(size_t)(Vb + kn),
                            1u, 32, HEAD_DIM, 32, HEAD_DIM, SEQ);
                __builtin_amdgcn_s_wait_tensorcnt(2);
            } else {
                __builtin_amdgcn_s_wait_tensorcnt(0);
            }
        }
        __syncthreads();
#else
        __syncthreads();
        stage_tile(Ktile[p], Kb + (size_t)kb * HEAD_DIM, 32, HEAD_DIM, HEAD_DIM, tid);
        stage_tile(Vtile[p], Vb + kb, HEAD_DIM, 32, SEQ, tid);
        __syncthreads();
#endif
        if (kb <= qbase + 15) {   // wave-uniform causal skip
            v8f s0 = zero8(), s1 = zero8();
#pragma unroll
            for (int c = 0; c < 4; ++c) {
                v16bf k0 = load_frag(Ktile[p], HEAD_DIM, 0,  32 * c, lane);
                v16bf k1 = load_frag(Ktile[p], HEAD_DIM, 16, 32 * c, lane);
                s0 = wmma_bf16(qf[c], k0, s0);
                s1 = wmma_bf16(qf[c], k1, s1);
            }
            float pmx[8];
            if (kb + 31 <= qbase) {          // interior tile: no masking needed
#pragma unroll
                for (int r = 0; r < 8; ++r) pmx[r] = fmaxf(s0[r], s1[r]);
            } else {                         // diagonal tile: causal mask
                const int key0 = kb + ln, key1 = kb + 16 + ln;
#pragma unroll
                for (int r = 0; r < 8; ++r) {
                    int row = qbase + r + 8 * hl;
                    float a = (key0 > row) ? -3e38f : s0[r];
                    float b = (key1 > row) ? -3e38f : s1[r];
                    s0[r] = a; s1[r] = b;
                    pmx[r] = fmaxf(a, b);
                }
            }
#pragma unroll
            for (int off = 1; off < 16; off <<= 1)
#pragma unroll
                for (int r = 0; r < 8; ++r)
                    pmx[r] = fmaxf(pmx[r], __shfl_xor(pmx[r], off, 32));

            float alpha[8];
#pragma unroll
            for (int r = 0; r < 8; ++r) {
                float mn = fmaxf(mrow[r], pmx[r]);
                alpha[r] = __expf(mrow[r] - mn);
                mrow[r] = mn;
            }
            float rs[8];
#pragma unroll
            for (int r = 0; r < 8; ++r) {
                float p0 = __expf(s0[r] - mrow[r]);
                float p1 = __expf(s1[r] - mrow[r]);
                s0[r] = p0; s1[r] = p1;
                rs[r] = p0 + p1;
            }
#pragma unroll
            for (int off = 1; off < 16; off <<= 1)
#pragma unroll
                for (int r = 0; r < 8; ++r)
                    rs[r] += __shfl_xor(rs[r], off, 32);
#pragma unroll
            for (int r = 0; r < 8; ++r) lrow[r] = lrow[r] * alpha[r] + rs[r];
#pragma unroll
            for (int c = 0; c < 8; ++c)
#pragma unroll
                for (int r = 0; r < 8; ++r) o[c][r] *= alpha[r];

            // P: C-layout f32 -> A-layout bf16 via wave-private LDS bounce
#pragma unroll
            for (int r = 0; r < 8; ++r) {
                pbuf[w][(r + 8 * hl) * 32 + ln]      = (__bf16)s0[r];
                pbuf[w][(r + 8 * hl) * 32 + 16 + ln] = (__bf16)s1[r];
            }
            // A-layout readback == generic fragment pattern (two ds_load_b128)
            v16bf pf = load_frag(pbuf[w], 32, 0, 0, lane);

#pragma unroll
            for (int c = 0; c < 8; ++c) {
                v16bf vf = load_frag(Vtile[p], 32, c * 16, 0, lane);
                o[c] = wmma_bf16(pf, vf, o[c]);
            }
        }
        __syncthreads();
    }

    const int b = bh >> 4, h = bh & (N_HEADS - 1);
#pragma unroll
    for (int c = 0; c < 8; ++c)
#pragma unroll
        for (int r = 0; r < 8; ++r) {
            int row = qbase + r + 8 * hl;
            float val = o[c][r] / lrow[r];
            size_t idx = ((size_t)b * SEQ + row) * D_MODEL + h * HEAD_DIM + c * 16 + ln;
            Aout[idx] = (__bf16)val;
        }
}

// ---------------------------------------------------------------------------
// Launch
// ---------------------------------------------------------------------------
extern "C" void kernel_launch(void* const* d_in, const int* in_sizes, int n_in,
                              void* d_out, int out_size, void* d_ws, size_t ws_size,
                              hipStream_t stream) {
    const float* x      = (const float*)d_in[0];  // (4,2048,2048)
    const float* qkv_w  = (const float*)d_in[1];  // (6144,2048)
    const float* wo     = (const float*)d_in[2];  // (2048,2048)
    const float* cosc   = (const float*)d_in[3];  // (2048,32)
    const float* sinc   = (const float*)d_in[4];  // (2048,32)
    float*       out    = (float*)d_out;          // (4,2048,2048)

    char* ws = (char*)d_ws;
    __bf16* qkvw_bf = (__bf16*)ws;  ws += (size_t)QKV_N * D_MODEL * 2;
    __bf16* wo_bf   = (__bf16*)ws;  ws += (size_t)D_MODEL * D_MODEL * 2;
    __bf16* qkv_bf  = (__bf16*)ws;  ws += (size_t)3 * ROWS * D_MODEL * 2;
    __bf16* q_bf    = (__bf16*)ws;  ws += (size_t)ROWS * D_MODEL * 2;
    __bf16* k_bf    = (__bf16*)ws;  ws += (size_t)ROWS * D_MODEL * 2;
    __bf16* vt_bf   = (__bf16*)ws;  ws += (size_t)ROWS * D_MODEL * 2;
    __bf16* attn_bf = (__bf16*)ws;  ws += (size_t)ROWS * D_MODEL * 2;

    cvt_f32_to_bf16<<<4096, 256, 0, stream>>>(qkv_w, qkvw_bf, QKV_N * D_MODEL);
    cvt_f32_to_bf16<<<2048, 256, 0, stream>>>(wo, wo_bf, D_MODEL * D_MODEL);

    gemm_wmma<float, 0><<<dim3(QKV_N / 256, ROWS / 64), 256, 0, stream>>>(
        x, qkvw_bf, (void*)qkv_bf, ROWS, QKV_N, D_MODEL);

    rope_split_kernel<<<dim3(BATCH * N_HEADS, SEQ), HEAD_DIM, 0, stream>>>(
        qkv_bf, cosc, sinc, q_bf, k_bf, vt_bf);

    flash_attn_kernel<<<dim3(BATCH * N_HEADS, SEQ / 128), 256, 0, stream>>>(
        q_bf, k_bf, vt_bf, attn_bf);

    gemm_wmma<__bf16, 1><<<dim3(D_MODEL / 256, ROWS / 64), 256, 0, stream>>>(
        attn_bf, wo_bf, (void*)out, ROWS, D_MODEL, D_MODEL);
}
